// MolAttentiveFP_54150947668224
// MI455X (gfx1250) — matrix operand order repaired and checked by reading, not compiled, not run on hardware
//
#include <hip/hip_runtime.h>
#include <hip/hip_bf16.h>
#include <math.h>

// ---------------------------------------------------------------------------
// AttentiveFP forward for MI455X (gfx1250, wave32).
// Dense GEMMs: bf16 WMMA (v_wmma_f32_16x16x32_bf16), fp32 accumulate.
// K <= 128 for every GEMM here, so each block stages its whole A (128xK) and
// B (64xK) operand into LDS ONCE via global_load_async_to_lds_b128 (ASYNCcnt),
// then runs all K-steps (16 WMMAs) fed purely by ds_load_b128.
// ---------------------------------------------------------------------------

#define HDIM   128
#define HDIM3  384
#define INCH   64
#define KPAD   8      // 16B row padding -> ds bank rotation

typedef __attribute__((ext_vector_type(16))) __bf16         bf16x16;
typedef __attribute__((ext_vector_type(8)))  float          v8f;
typedef __attribute__((ext_vector_type(8)))  unsigned short us8v;

union BF16Frag { unsigned short u[16]; bf16x16 v; };

__device__ __forceinline__ unsigned short f2bf(float f) {
  unsigned int u = __float_as_uint(f);
  unsigned int r = u + 0x7FFFu + ((u >> 16) & 1u);   // round-to-nearest-even
  return (unsigned short)(r >> 16);
}
__device__ __forceinline__ float lrelu01(float v) { return v > 0.f ? v : 0.01f * v; }
__device__ __forceinline__ float sigmf(float v)   { return 1.f / (1.f + __expf(-v)); }
__device__ __forceinline__ void atomicMaxF(float* a, float v) {
  if (v >= 0.f) atomicMax((int*)a, __float_as_int(v));
  else          atomicMin((unsigned int*)a, __float_as_uint(v));
}

// async memory -> LDS copy of 16 bytes (8 bf16). ldsp is a flat pointer to
// __shared__; its low 32 bits are the wave-relative LDS byte address.
__device__ __forceinline__ void async_b128_to_lds(const unsigned short* gp,
                                                  unsigned short* ldsp) {
  const unsigned      lo = (unsigned)(size_t)ldsp;
  const unsigned long long ga = (unsigned long long)(size_t)gp;
  asm volatile("global_load_async_to_lds_b128 %0, %1, off"
               :: "v"(lo), "v"(ga) : "memory");
}
__device__ __forceinline__ void wait_async_then_barrier() {
  asm volatile("s_wait_asynccnt 0x0" ::: "memory");
  __syncthreads();
}

// ---------------------------------------------------------------------------
// WMMA GEMM:  Y[nrows, M] = act( A[nrows, K] @ W[M, K]^T + bias )
// A, W packed bf16. block = 256 threads (8 waves), tile 128 rows x 64 cols.
// Wave w: rows [16w,16w+16) x 64 cols (4 accumulators, 4 WMMAs per K-step).
// act: 0 none, 1 leaky_relu(0.01), 2 relu. Optional bf16 shadow output Ybf.
// Requires K in {64,128}, M % 64 == 0.
// ---------------------------------------------------------------------------
__global__ __launch_bounds__(256)
void k_gemm_wmma(const unsigned short* __restrict__ Abf, int nrows, int K,
                 const unsigned short* __restrict__ Wbf,
                 const float* __restrict__ bias, int bias_stride,
                 float* __restrict__ Y, unsigned short* __restrict__ Ybf,
                 int M, int act)
{
  __shared__ alignas(16) unsigned short Alds[128 * (HDIM + KPAD)];  // 34 KB
  __shared__ alignas(16) unsigned short Blds[64  * (HDIM + KPAD)];  // 17 KB

  const int tid    = threadIdx.x;
  const int wave   = tid >> 5;
  const int lane   = tid & 31;
  const int rowblk = blockIdx.x * 128;
  const int colblk = blockIdx.y * 64;
  const int KP     = K + KPAD;

  const int hi   = (lane >= 16) ? 1 : 0;
  const int aoff = hi ? 8  : 0;   // A frag K offset per ISA 16-bit A 16x32 layout
  const int boff = hi ? 8  : 0;   // B frag K offset per half-wave (same table order as A)
  const int arow = (lane & 15) + wave * 16;
  const int bcol = lane & 15;

  v8f acc[4];
  #pragma unroll
  for (int j = 0; j < 4; ++j)
    #pragma unroll
    for (int i = 0; i < 8; ++i) acc[j][i] = 0.f;

  us8v zz;
  #pragma unroll
  for (int i = 0; i < 8; ++i) zz[i] = 0;

  // ---- stage the entire A block [128 x K] (async mem->LDS, no VGPR trip)
  {
    const int r  = tid >> 1;
    const int c0 = (tid & 1) * (K >> 1);
    const int grow = rowblk + r;
    if (grow < nrows) {
      for (int c = 0; c < (K >> 1); c += 8)
        async_b128_to_lds(Abf + (size_t)grow * K + c0 + c, &Alds[r * KP + c0 + c]);
    } else {
      for (int c = 0; c < (K >> 1); c += 8)
        *(us8v*)&Alds[r * KP + c0 + c] = zz;
    }
  }
  // ---- stage the entire W tile [64 x K]
  {
    const int wr = tid >> 2;
    const int c0 = (tid & 3) * (K >> 2);
    for (int c = 0; c < (K >> 2); c += 8)
      async_b128_to_lds(Wbf + (size_t)(colblk + wr) * K + c0 + c,
                        &Blds[wr * KP + c0 + c]);
  }
  wait_async_then_barrier();

  // ---- all K-steps from LDS: 2 + 8 ds_load_b128 and 4 WMMAs per step
  for (int k0 = 0; k0 < K; k0 += 32) {
    BF16Frag af;
    *(us8v*)&af.u[0] = *(const us8v*)&Alds[arow * KP + k0 + aoff];
    *(us8v*)&af.u[8] = *(const us8v*)&Alds[arow * KP + k0 + 16 + aoff];
    #pragma unroll
    for (int j = 0; j < 4; ++j) {
      BF16Frag bfr;
      *(us8v*)&bfr.u[0] = *(const us8v*)&Blds[(j * 16 + bcol) * KP + k0 + boff];
      *(us8v*)&bfr.u[8] = *(const us8v*)&Blds[(j * 16 + bcol) * KP + k0 + 16 + boff];
      acc[j] = __builtin_amdgcn_wmma_f32_16x16x32_bf16(
          false, af.v, false, bfr.v, (short)0, acc[j], false, false);
    }
  }

  // ---- epilogue: C layout VGPR r -> (M = r + 8*hi, N = lane&15)
  const int rbase = rowblk + wave * 16 + (hi ? 8 : 0);
  #pragma unroll
  for (int j = 0; j < 4; ++j) {
    const int ccol = colblk + j * 16 + bcol;
    const float bv = bias ? bias[(size_t)ccol * bias_stride] : 0.f;
    #pragma unroll
    for (int r = 0; r < 8; ++r) {
      const int grow = rbase + r;
      if (grow < nrows) {
        float v = acc[j][r] + bv;
        if (act == 1)      v = lrelu01(v);
        else if (act == 2) v = fmaxf(v, 0.f);
        Y[(size_t)grow * M + ccol] = v;
        if (Ybf) Ybf[(size_t)grow * M + ccol] = f2bf(v);
      }
    }
  }
}

// fp32 -> bf16, packed
__global__ __launch_bounds__(256)
void k_f2bf(const float* __restrict__ a, unsigned short* __restrict__ b, long n)
{
  const long i = (long)blockIdx.x * 256 + threadIdx.x;
  if (i < n) b[i] = f2bf(a[i]);
}

// strided weight fp32 [M, ldw] -> packed bf16 [M, K]
__global__ __launch_bounds__(256)
void k_w2bf(const float* __restrict__ W, int ldw, int M, int K,
            unsigned short* __restrict__ out)
{
  const long i = (long)blockIdx.x * 256 + threadIdx.x;
  if (i >= (long)M * K) return;
  const int m = (int)(i / K), k = (int)(i % K);
  out[i] = f2bf(W[(size_t)m * ldw + k]);
}

// wave-per-row dot: y[n] = dot(X[n, 0:128], v)
__global__ __launch_bounds__(256)
void k_rowdot(const float* __restrict__ X, const float* __restrict__ v,
              float* __restrict__ y, int nrows)
{
  const int n    = blockIdx.x * 8 + (threadIdx.x >> 5);
  const int lane = threadIdx.x & 31;
  if (n >= nrows) return;
  const float4 xv = *(const float4*)(X + (size_t)n * HDIM + lane * 4);
  const float4 vv = *(const float4*)(v + lane * 4);
  float s = xv.x * vv.x + xv.y * vv.y + xv.z * vv.z + xv.w * vv.w;
  #pragma unroll
  for (int off = 16; off; off >>= 1) s += __shfl_down(s, off);
  if (lane == 0) y[n] = s;
}

__global__ __launch_bounds__(256)
void k_fill(float* __restrict__ p, float val, long n)
{
  const long i = (long)blockIdx.x * 256 + threadIdx.x;
  if (i < n) p[i] = val;
}

// edge pass 1: logit + per-dst float max
__global__ __launch_bounds__(256)
void k_edge_logit(const int* __restrict__ src, const int* __restrict__ dst,
                  const float* __restrict__ aj, const float* __restrict__ ai,
                  float* __restrict__ logit, float* __restrict__ m, int E)
{
  const int e = blockIdx.x * 256 + threadIdx.x;
  if (e >= E) return;
  const int s = src[e], d = dst[e];
  const float l = lrelu01(aj[s] + ai[d]);
  logit[e] = l;
  atomicMaxF(&m[d], l);
}

// edge pass 2: ev = exp(logit - m[dst]); s[dst]+=ev; hacc[dst]+=ev*feat[src]
__global__ __launch_bounds__(256)
void k_edge_accum(const int* __restrict__ src, const int* __restrict__ dst,
                  const float* __restrict__ logit, const float* __restrict__ m,
                  float* __restrict__ ssum, const float* __restrict__ feat,
                  float* __restrict__ hacc, int E)
{
  const int e    = blockIdx.x * 8 + (threadIdx.x >> 5);
  const int lane = threadIdx.x & 31;
  if (e >= E) return;
  const int s = src[e], d = dst[e];
  __builtin_prefetch(feat + (size_t)s * HDIM, 0, 0);   // global_prefetch_b8
  const float ev = __expf(logit[e] - m[d]);
  if (lane == 0) atomicAdd(&ssum[d], ev);
  const float4 f = *(const float4*)(feat + (size_t)s * HDIM + lane * 4);
  float* hp = hacc + (size_t)d * HDIM + lane * 4;
  atomicAdd(hp + 0, ev * f.x);
  atomicAdd(hp + 1, ev * f.y);
  atomicAdd(hp + 2, ev * f.z);
  atomicAdd(hp + 3, ev * f.w);
}

// node-level variants (segment = batch)
__global__ __launch_bounds__(256)
void k_node_logit(const int* __restrict__ batch, const float* __restrict__ asrc,
                  const float* __restrict__ adst, float* __restrict__ logit,
                  float* __restrict__ m, int N)
{
  const int n = blockIdx.x * 256 + threadIdx.x;
  if (n >= N) return;
  const int g = batch[n];
  const float l = lrelu01(asrc[n] + adst[g]);
  logit[n] = l;
  atomicMaxF(&m[g], l);
}

__global__ __launch_bounds__(256)
void k_node_accum(const int* __restrict__ batch, const float* __restrict__ logit,
                  const float* __restrict__ m, float* __restrict__ ssum,
                  const float* __restrict__ feat, float* __restrict__ hacc, int N)
{
  const int n    = blockIdx.x * 8 + (threadIdx.x >> 5);
  const int lane = threadIdx.x & 31;
  if (n >= N) return;
  const int g = batch[n];
  const float ev = __expf(logit[n] - m[g]);
  if (lane == 0) atomicAdd(&ssum[g], ev);
  const float4 f = *(const float4*)(feat + (size_t)n * HDIM + lane * 4);
  float* hp = hacc + (size_t)g * HDIM + lane * 4;
  atomicAdd(hp + 0, ev * f.x);
  atomicAdd(hp + 1, ev * f.y);
  atomicAdd(hp + 2, ev * f.z);
  atomicAdd(hp + 3, ev * f.w);
}

// h = elu(hacc / (s + 1e-16) + bias)  (in-place safe) + bf16 shadow
__global__ __launch_bounds__(256)
void k_seg_finish(const float* __restrict__ hacc, const float* __restrict__ ssum,
                  const float* __restrict__ bias, float* __restrict__ out,
                  unsigned short* __restrict__ out_bf, int nrows)
{
  const long i = (long)blockIdx.x * 256 + threadIdx.x;
  if (i >= (long)nrows * HDIM) return;
  const int n = (int)(i >> 7), f = (int)(i & 127);
  float v = hacc[i] / (ssum[n] + 1e-16f) + bias[f];
  v = v > 0.f ? v : __expf(v) - 1.f;
  out[i] = v;
  out_bf[i] = f2bf(v);
}

// GRU pointwise: out = relu((1-z)*n + z*h) + bf16 shadow (in-place safe)
__global__ __launch_bounds__(256)
void k_gru(const float* __restrict__ gi, const float* __restrict__ gh,
           const float* __restrict__ hstate, float* __restrict__ out,
           unsigned short* __restrict__ out_bf, int nrows)
{
  const long i = (long)blockIdx.x * 256 + threadIdx.x;
  if (i >= (long)nrows * HDIM) return;
  const int n = (int)(i >> 7), f = (int)(i & 127);
  const size_t b = (size_t)n * HDIM3 + f;
  const float ir = gi[b], iz = gi[b + HDIM], in_ = gi[b + 2 * HDIM];
  const float hr = gh[b], hz = gh[b + HDIM], hn  = gh[b + 2 * HDIM];
  const float r = sigmf(ir + hr);
  const float z = sigmf(iz + hz);
  const float nn = tanhf(in_ + r * hn);
  float v = (1.f - z) * nn + z * hstate[i];
  v = fmaxf(v, 0.f);
  out[i] = v;
  out_bf[i] = f2bf(v);
}

// scatter node features into graph accumulator
__global__ __launch_bounds__(256)
void k_scatter_nodes(const float* __restrict__ xc, const int* __restrict__ batch,
                     float* __restrict__ acc, int N)
{
  const int n    = blockIdx.x * 8 + (threadIdx.x >> 5);
  const int lane = threadIdx.x & 31;
  if (n >= N) return;
  const int g = batch[n];
  const float4 f = *(const float4*)(xc + (size_t)n * HDIM + lane * 4);
  float* p = acc + (size_t)g * HDIM + lane * 4;
  atomicAdd(p + 0, f.x);
  atomicAdd(p + 1, f.y);
  atomicAdd(p + 2, f.z);
  atomicAdd(p + 3, f.w);
}

__global__ __launch_bounds__(256)
void k_relu_copy(const float* __restrict__ a, float* __restrict__ b,
                 unsigned short* __restrict__ b_bf, long n)
{
  const long i = (long)blockIdx.x * 256 + threadIdx.x;
  if (i >= n) return;
  const float v = fmaxf(a[i], 0.f);
  b[i] = v;
  b_bf[i] = f2bf(v);
}

// final: y[g] = dot(out[g], lin2_W) + lin2_b
__global__ __launch_bounds__(256)
void k_lin2(const float* __restrict__ out, const float* __restrict__ W,
            const float* __restrict__ b, float* __restrict__ y, int G)
{
  const int g    = blockIdx.x * 8 + (threadIdx.x >> 5);
  const int lane = threadIdx.x & 31;
  if (g >= G) return;
  const float4 o = *(const float4*)(out + (size_t)g * HDIM + lane * 4);
  const float4 w = *(const float4*)(W + lane * 4);
  float s = o.x * w.x + o.y * w.y + o.z * w.z + o.w * w.w;
  #pragma unroll
  for (int off = 16; off; off >>= 1) s += __shfl_down(s, off);
  if (lane == 0) y[g] = s + b[0];
}

// ---------------------------------------------------------------------------
// Host orchestration
// ---------------------------------------------------------------------------
static inline int cdiv(long a, long b) { return (int)((a + b - 1) / b); }

extern "C" void kernel_launch(void* const* d_in, const int* in_sizes, int n_in,
                              void* d_out, int out_size, void* d_ws, size_t ws_size,
                              hipStream_t stream)
{
  const int N = in_sizes[0] / INCH;
  const int E = in_sizes[1] / 2;
  const int G = out_size;            // OUT == 1
  const int L = 2;                   // NUM_LAYERS - 1
  const int T = 2;                   // NUM_TIMESTEPS

  const float* x        = (const float*)d_in[0];
  const int*   src      = (const int*)d_in[1];
  const int*   dst      = src + E;
  const int*   batch    = (const int*)d_in[2];
  const float* lin1_W   = (const float*)d_in[3];
  const float* lin1_b   = (const float*)d_in[4];
  const float* gate_W1  = (const float*)d_in[5];   // [H, H+1]
  const float* gate_W2  = (const float*)d_in[6];
  const float* gate_al  = (const float*)d_in[7];
  const float* gate_ar  = (const float*)d_in[8];
  const float* gate_b   = (const float*)d_in[9];
  const float* g0_Wi    = (const float*)d_in[10];
  const float* g0_bi    = (const float*)d_in[11];
  const float* g0_Wh    = (const float*)d_in[12];
  const float* g0_bh    = (const float*)d_in[13];
  const float* atom_W   = (const float*)d_in[14];
  const float* atom_as  = (const float*)d_in[15];
  const float* atom_ad  = (const float*)d_in[16];
  const float* atom_b   = (const float*)d_in[17];
  const float* a_gWi    = (const float*)d_in[18];
  const float* a_gbi    = (const float*)d_in[19];
  const float* a_gWh    = (const float*)d_in[20];
  const float* a_gbh    = (const float*)d_in[21];
  const float* mol_W    = (const float*)d_in[22];
  const float* mol_as   = (const float*)d_in[23];
  const float* mol_ad   = (const float*)d_in[24];
  const float* mol_b    = (const float*)d_in[25];
  const float* m_gWi    = (const float*)d_in[26];
  const float* m_gbi    = (const float*)d_in[27];
  const float* m_gWh    = (const float*)d_in[28];
  const float* m_gbh    = (const float*)d_in[29];
  const float* lin2_W   = (const float*)d_in[30];
  const float* lin2_b   = (const float*)d_in[31];
  float* y_out = (float*)d_out;

  // ---- workspace carve-up ----
  float* ws = (float*)d_ws;
  size_t o = 0;
  auto take = [&](size_t nf) { float* p = ws + o; o += (nf + 63) & ~(size_t)63; return p; };
  auto take_us = [&](size_t ns) { return (unsigned short*)take((ns + 1) / 2); };

  float* x0    = take((size_t)N * HDIM);
  float* featN = take((size_t)N * HDIM);   // gate t -> xw2 -> atom xw -> mol xw_src
  float* hmsg  = take((size_t)N * HDIM);
  float* xc    = take((size_t)N * HDIM);
  float* gi    = take((size_t)N * HDIM3);
  float* gh    = take((size_t)N * HDIM3);
  float* ajn   = take(N);
  float* ain   = take(N);
  float* mN    = take(N);
  float* sN    = take(N);
  float* asrcN = take(N);
  float* logN  = take(N);
  float* logE  = take(E);
  float* outG  = take((size_t)G * HDIM);
  float* accG  = take((size_t)G * HDIM);
  float* outW  = take((size_t)G * HDIM);
  float* giG   = take((size_t)G * HDIM3);
  float* ghG   = take((size_t)G * HDIM3);
  float* adstG = take(G);
  float* mG    = take(G);
  float* sG    = take(G);
  // bf16 shadows of GEMM A-operands
  unsigned short* xbf    = take_us((size_t)N * INCH);
  unsigned short* x0bf   = take_us((size_t)N * HDIM);
  unsigned short* hmsgbf = take_us((size_t)N * HDIM);
  unsigned short* xcbf   = take_us((size_t)N * HDIM);
  unsigned short* accGbf = take_us((size_t)G * HDIM);
  unsigned short* outGbf = take_us((size_t)G * HDIM);
  (void)ws_size; (void)n_in;

  const float NEG_INF = -__builtin_huge_valf();
  dim3 blk(256);

  // ---- convert all weights to packed bf16 once ----
  struct WC { const float* W; int ldw, M, K; unsigned short* dst; };
  unsigned short* wlin1 = take_us((size_t)HDIM * INCH);
  unsigned short* wg1   = take_us((size_t)HDIM * HDIM);
  unsigned short* wg2   = take_us((size_t)HDIM * HDIM);
  unsigned short* wg0i  = take_us((size_t)HDIM3 * HDIM);
  unsigned short* wg0h  = take_us((size_t)HDIM3 * HDIM);
  unsigned short* watm0 = take_us((size_t)HDIM * HDIM);
  unsigned short* watm1 = take_us((size_t)HDIM * HDIM);
  unsigned short* wai0  = take_us((size_t)HDIM3 * HDIM);
  unsigned short* wai1  = take_us((size_t)HDIM3 * HDIM);
  unsigned short* wah0  = take_us((size_t)HDIM3 * HDIM);
  unsigned short* wah1  = take_us((size_t)HDIM3 * HDIM);
  unsigned short* wmol  = take_us((size_t)HDIM * HDIM);
  unsigned short* wmi   = take_us((size_t)HDIM3 * HDIM);
  unsigned short* wmh   = take_us((size_t)HDIM3 * HDIM);
  const WC wconv[] = {
    { lin1_W,  INCH,     HDIM,  INCH, wlin1 },
    { gate_W1, HDIM + 1, HDIM,  HDIM, wg1   },   // drop bias column
    { gate_W2, HDIM,     HDIM,  HDIM, wg2   },
    { g0_Wi,   HDIM,     HDIM3, HDIM, wg0i  },
    { g0_Wh,   HDIM,     HDIM3, HDIM, wg0h  },
    { atom_W,                HDIM, HDIM,  HDIM, watm0 },
    { atom_W + HDIM * HDIM,  HDIM, HDIM,  HDIM, watm1 },
    { a_gWi,                 HDIM, HDIM3, HDIM, wai0  },
    { a_gWi + HDIM3 * HDIM,  HDIM, HDIM3, HDIM, wai1  },
    { a_gWh,                 HDIM, HDIM3, HDIM, wah0  },
    { a_gWh + HDIM3 * HDIM,  HDIM, HDIM3, HDIM, wah1  },
    { mol_W,   HDIM, HDIM,  HDIM, wmol },
    { m_gWi,   HDIM, HDIM3, HDIM, wmi  },
    { m_gWh,   HDIM, HDIM3, HDIM, wmh  },
  };
  for (const WC& wc : wconv)
    k_w2bf<<<cdiv((long)wc.M * wc.K, 256), blk, 0, stream>>>(wc.W, wc.ldw, wc.M, wc.K, wc.dst);
  k_f2bf<<<cdiv((long)N * INCH, 256), blk, 0, stream>>>(x, xbf, (long)N * INCH);

  auto gemm = [&](const unsigned short* A, int rows, int K, const unsigned short* W,
                  const float* bias, int bstr, float* Y, unsigned short* Ybf,
                  int M, int act) {
    dim3 grid(cdiv(rows, 128), M / 64);
    k_gemm_wmma<<<grid, blk, 0, stream>>>(A, rows, K, W, bias, bstr, Y, Ybf, M, act);
  };

  // ---- lin1: x0 = lrelu(x @ lin1_W.T + lin1_b)  (+ bf16 shadow)
  gemm(xbf, N, INCH, wlin1, lin1_b, 1, x0, x0bf, HDIM, 1);

  // ---- GATEConv
  gemm(x0bf, N, HDIM, wg1, gate_W1 + HDIM, HDIM + 1, featN, nullptr, HDIM, 1); // t
  k_rowdot<<<cdiv(N, 8), blk, 0, stream>>>(featN, gate_al, ajn, N);
  k_rowdot<<<cdiv(N, 8), blk, 0, stream>>>(x0,    gate_ar, ain, N);
  gemm(x0bf, N, HDIM, wg2, nullptr, 0, featN, nullptr, HDIM, 0);               // xw2
  k_fill<<<cdiv(N, 256), blk, 0, stream>>>(mN, NEG_INF, N);
  k_fill<<<cdiv(N, 256), blk, 0, stream>>>(sN, 0.f, N);
  k_fill<<<cdiv((long)N * HDIM, 256), blk, 0, stream>>>(hmsg, 0.f, (long)N * HDIM);
  k_edge_logit<<<cdiv(E, 256), blk, 0, stream>>>(src, dst, ajn, ain, logE, mN, E);
  k_edge_accum<<<cdiv(E, 8), blk, 0, stream>>>(src, dst, logE, mN, sN, featN, hmsg, E);
  k_seg_finish<<<cdiv((long)N * HDIM, 256), blk, 0, stream>>>(hmsg, sN, gate_b, hmsg, hmsgbf, N);
  // GRU0: xc = relu(gru(hmsg, x0))
  gemm(hmsgbf, N, HDIM, wg0i, g0_bi, 1, gi, nullptr, HDIM3, 0);
  gemm(x0bf,   N, HDIM, wg0h, g0_bh, 1, gh, nullptr, HDIM3, 0);
  k_gru<<<cdiv((long)N * HDIM, 256), blk, 0, stream>>>(gi, gh, x0, xc, xcbf, N);

  // ---- extra GATConv + GRU layers
  const unsigned short* watm[2] = { watm0, watm1 };
  const unsigned short* wai[2]  = { wai0,  wai1  };
  const unsigned short* wah[2]  = { wah0,  wah1  };
  for (int l = 0; l < L; ++l) {
    const float* asl = atom_as + (size_t)l * HDIM;
    const float* adl = atom_ad + (size_t)l * HDIM;
    const float* bl  = atom_b  + (size_t)l * HDIM;
    const float* bil = a_gbi   + (size_t)l * HDIM3;
    const float* bhl = a_gbh   + (size_t)l * HDIM3;
    gemm(xcbf, N, HDIM, watm[l], nullptr, 0, featN, nullptr, HDIM, 0);   // xw
    k_rowdot<<<cdiv(N, 8), blk, 0, stream>>>(featN, asl, ajn, N);
    k_rowdot<<<cdiv(N, 8), blk, 0, stream>>>(featN, adl, ain, N);
    k_fill<<<cdiv(N, 256), blk, 0, stream>>>(mN, NEG_INF, N);
    k_fill<<<cdiv(N, 256), blk, 0, stream>>>(sN, 0.f, N);
    k_fill<<<cdiv((long)N * HDIM, 256), blk, 0, stream>>>(hmsg, 0.f, (long)N * HDIM);
    k_edge_logit<<<cdiv(E, 256), blk, 0, stream>>>(src, dst, ajn, ain, logE, mN, E);
    k_edge_accum<<<cdiv(E, 8), blk, 0, stream>>>(src, dst, logE, mN, sN, featN, hmsg, E);
    k_seg_finish<<<cdiv((long)N * HDIM, 256), blk, 0, stream>>>(hmsg, sN, bl, hmsg, hmsgbf, N);
    gemm(hmsgbf, N, HDIM, wai[l], bil, 1, gi, nullptr, HDIM3, 0);
    gemm(xcbf,   N, HDIM, wah[l], bhl, 1, gh, nullptr, HDIM3, 0);
    k_gru<<<cdiv((long)N * HDIM, 256), blk, 0, stream>>>(gi, gh, xc, xc, xcbf, N); // in-place
  }

  // ---- attentive readout
  k_fill<<<cdiv((long)G * HDIM, 256), blk, 0, stream>>>(accG, 0.f, (long)G * HDIM);
  k_scatter_nodes<<<cdiv(N, 8), blk, 0, stream>>>(xc, batch, accG, N);
  k_relu_copy<<<cdiv((long)G * HDIM, 256), blk, 0, stream>>>(accG, outG, outGbf, (long)G * HDIM);
  gemm(xcbf, N, HDIM, wmol, nullptr, 0, featN, nullptr, HDIM, 0);  // xw_src
  k_rowdot<<<cdiv(N, 8), blk, 0, stream>>>(featN, mol_as, asrcN, N);

  for (int t = 0; t < T; ++t) {
    gemm(outGbf, G, HDIM, wmol, nullptr, 0, outW, nullptr, HDIM, 0);
    k_rowdot<<<cdiv(G, 8), blk, 0, stream>>>(outW, mol_ad, adstG, G);
    k_fill<<<cdiv(G, 256), blk, 0, stream>>>(mG, NEG_INF, G);
    k_fill<<<cdiv(G, 256), blk, 0, stream>>>(sG, 0.f, G);
    k_fill<<<cdiv((long)G * HDIM, 256), blk, 0, stream>>>(accG, 0.f, (long)G * HDIM);
    k_node_logit<<<cdiv(N, 256), blk, 0, stream>>>(batch, asrcN, adstG, logN, mG, N);
    k_node_accum<<<cdiv(N, 8), blk, 0, stream>>>(batch, logN, mG, sG, featN, accG, N);
    k_seg_finish<<<cdiv((long)G * HDIM, 256), blk, 0, stream>>>(accG, sG, mol_b, accG, accGbf, G);
    gemm(accGbf, G, HDIM, wmi, m_gbi, 1, giG, nullptr, HDIM3, 0);
    gemm(outGbf, G, HDIM, wmh, m_gbh, 1, ghG, nullptr, HDIM3, 0);
    k_gru<<<cdiv((long)G * HDIM, 256), blk, 0, stream>>>(giG, ghG, outG, outG, outGbf, G);
  }

  // ---- lin2
  k_lin2<<<cdiv(G, 8), blk, 0, stream>>>(outG, lin2_W, lin2_b, y_out, G);
}
